// GraphSAGENet_57801669869722
// MI455X (gfx1250) — matrix operand order repaired and checked by reading, not compiled, not run on hardware
//
#include <hip/hip_runtime.h>
#include <hip/hip_bf16.h>

typedef _Float16 v16h __attribute__((ext_vector_type(16)));
typedef _Float16 v8h  __attribute__((ext_vector_type(8)));
typedef float    v8f  __attribute__((ext_vector_type(8)));

#define TASKS 12
#define HID 128

// ---------------------------------------------------------------- utility
__global__ void k_cast_f16(const float* __restrict__ x, _Float16* __restrict__ o, int n) {
    int i = blockIdx.x * blockDim.x + threadIdx.x;
    if (i < n) o[i] = (_Float16)x[i];
}

__global__ void k_degree(const long long* __restrict__ dst, float* __restrict__ deg, int E) {
    int e = blockIdx.x * blockDim.x + threadIdx.x;
    if (e < E) atomicAdd(&deg[dst[e]], 1.0f);
}

__global__ void k_deginv(float* __restrict__ deg, int n) {
    int i = blockIdx.x * blockDim.x + threadIdx.x;
    if (i < n) deg[i] = 1.0f / fmaxf(deg[i], 1.0f);
}

// ------------------------------------------------- edge scatter (mean agg numerator)
// thread -> (edge, 4-feature chunk); f32 atomics land in L2 (agg fits in 192MB L2)
__global__ void k_scatter(const _Float16* __restrict__ h,
                          const long long* __restrict__ src,
                          const long long* __restrict__ dst,
                          float* __restrict__ agg,
                          int E, int K, int shift) {
    long long idx = (long long)blockIdx.x * blockDim.x + threadIdx.x;
    int chunks = K >> 2;
    if (idx >= (long long)E * chunks) return;
    int e = (int)(idx >> shift);
    int c = (int)(idx & (chunks - 1)) << 2;
    long long s = src[e], d = dst[e];
    const _Float16* hp = h + s * (long long)K + c;
    float* ap = agg + d * (long long)K + c;
#pragma unroll
    for (int j = 0; j < 4; ++j) atomicAdd(&ap[j], (float)hp[j]);
}

// ------------------------------------------------- fused SAGE layer GEMM (WMMA)
// hout[16*mt+r, :] = relu(h @ Wl^T + bl + (agg*deg_inv) @ Wr^T), out width fixed 128
// Wl/Wr are pre-converted f16, row-major [128, K] -> B fragment is one v16h load.
template <int K>
__global__ __launch_bounds__(256) void k_sage_gemm(
    const _Float16* __restrict__ h, const float* __restrict__ agg,
    const float* __restrict__ dinv,
    const _Float16* __restrict__ Wl, const float* __restrict__ bl,
    const _Float16* __restrict__ Wr, _Float16* __restrict__ hout, int mtiles) {
    constexpr int KCH = K / 32;
    int wave = threadIdx.x >> 5;
    int lane = threadIdx.x & 31;
    int mt = blockIdx.x * 8 + wave;
    if (mt >= mtiles) return;                  // uniform per wave -> EXEC all-1 inside
    int m = lane & 15, half = lane >> 4;
    long long row = (long long)mt * 16 + m;

    // A fragments: contiguous k order (same as B) -> lane (m,half) holds row m,
    // K = [kc*32 + 16*half, +16) in elems 0..15
    v16h Ah[KCH], Ag[KCH];
    float di = dinv[row];
    const _Float16* hrow = h + row * K;
    const float* arow = agg + row * K;
#pragma unroll
    for (int kc = 0; kc < KCH; ++kc) {
        int b1 = kc * 32 + half * 16;
        int b2 = b1 + 8;
        v8h lo = *(const v8h*)(hrow + b1);
        v8h hi = *(const v8h*)(hrow + b2);
        v16h a;
#pragma unroll
        for (int j = 0; j < 8; ++j) { a[j] = lo[j]; a[j + 8] = hi[j]; }
        Ah[kc] = a;
        v16h g;
#pragma unroll
        for (int j = 0; j < 8; ++j) {
            g[j]     = (_Float16)(arow[b1 + j] * di);
            g[j + 8] = (_Float16)(arow[b2 + j] * di);
        }
        Ag[kc] = g;
    }

#pragma unroll
    for (int nt = 0; nt < 8; ++nt) {
        int col = nt * 16 + m;
        float bias = bl[col];
        v8f acc;
#pragma unroll
        for (int r = 0; r < 8; ++r) acc[r] = bias;
        // B layout: lane (n=m, half) holds B[k, n] = W[n, k] for k in [kb, kb+16)
#pragma unroll
        for (int kc = 0; kc < KCH; ++kc) {
            int kb = kc * 32 + half * 16;
            v16h B = *(const v16h*)(Wl + (long long)col * K + kb);
            acc = __builtin_amdgcn_wmma_f32_16x16x32_f16(false, Ah[kc], false, B,
                                                         (short)0, acc, false, false);
        }
#pragma unroll
        for (int kc = 0; kc < KCH; ++kc) {
            int kb = kc * 32 + half * 16;
            v16h B = *(const v16h*)(Wr + (long long)col * K + kb);
            acc = __builtin_amdgcn_wmma_f32_16x16x32_f16(false, Ag[kc], false, B,
                                                         (short)0, acc, false, false);
        }
        // D layout: VGPR r -> row (8*half + r), col = nt*16 + m
#pragma unroll
        for (int r = 0; r < 8; ++r) {
            long long orow = (long long)mt * 16 + half * 8 + r;
            float v = acc[r];
            v = v > 0.0f ? v : 0.0f;
            hout[orow * HID + col] = (_Float16)v;
        }
    }
}

// ------------------------------------------------- global mean pool + head
__global__ void k_pool(const _Float16* __restrict__ h, const long long* __restrict__ batch,
                       float* __restrict__ pooled, int Nn) {
    long long idx = (long long)blockIdx.x * blockDim.x + threadIdx.x;
    if (idx >= (long long)Nn * 32) return;
    int node = (int)(idx >> 5);
    int c = (int)(idx & 31) << 2;
    long long g = batch[node];
    const _Float16* hp = h + (long long)node * HID + c;
    float* pp = pooled + g * HID + c;
#pragma unroll
    for (int j = 0; j < 4; ++j) atomicAdd(&pp[j], (float)hp[j]);
}

__global__ void k_cnt(const long long* __restrict__ batch, float* __restrict__ cnt, int Nn) {
    int i = blockIdx.x * blockDim.x + threadIdx.x;
    if (i < Nn) atomicAdd(&cnt[batch[i]], 1.0f);
}

__global__ void k_out(const float* __restrict__ pooled, const float* __restrict__ cnt,
                      const float* __restrict__ Wo, const float* __restrict__ bo,
                      float* __restrict__ out, int G) {
    int idx = blockIdx.x * blockDim.x + threadIdx.x;
    if (idx >= G * TASKS) return;
    int g = idx / TASKS, t = idx % TASKS;
    float ic = 1.0f / fmaxf(cnt[g], 1.0f);
    float s = 0.0f;
    const float* pp = pooled + (long long)g * HID;
    const float* wp = Wo + (long long)t * HID;
#pragma unroll 4
    for (int k = 0; k < HID; ++k) s += pp[k] * wp[k];
    out[idx] = s * ic + bo[t];
}

// ---------------------------------------------------------------- launch
extern "C" void kernel_launch(void* const* d_in, const int* in_sizes, int n_in,
                              void* d_out, int out_size, void* d_ws, size_t ws_size,
                              hipStream_t stream) {
    const float*     x     = (const float*)d_in[0];
    const long long* ei    = (const long long*)d_in[1];
    const long long* batch = (const long long*)d_in[2];
    const float* Wl1 = (const float*)d_in[3];
    const float* bl1 = (const float*)d_in[4];
    const float* Wr1 = (const float*)d_in[5];
    const float* Wl2 = (const float*)d_in[6];
    const float* bl2 = (const float*)d_in[7];
    const float* Wr2 = (const float*)d_in[8];
    const float* Wl3 = (const float*)d_in[9];
    const float* bl3 = (const float*)d_in[10];
    const float* Wr3 = (const float*)d_in[11];
    const float* Wo  = (const float*)d_in[12];
    const float* bo  = (const float*)d_in[13];
    float* out = (float*)d_out;

    const int Nn = in_sizes[0] / 64;   // 100000
    const int E  = in_sizes[1] / 2;    // 1250000
    const int G  = out_size / TASKS;   // 2048
    const long long* src = ei;
    const long long* dst = ei + E;

    char* p = (char*)d_ws;
    auto carve = [&](size_t bytes) -> void* {
        void* r = (void*)p;
        p += (bytes + 255) & ~(size_t)255;
        return r;
    };
    float*     deg    = (float*)carve((size_t)Nn * 4);            // -> deg_inv in place
    _Float16*  hA     = (_Float16*)carve((size_t)Nn * HID * 2);
    _Float16*  hB     = (_Float16*)carve((size_t)Nn * HID * 2);
    float*     agg    = (float*)carve((size_t)Nn * HID * 4);
    float*     pooled = (float*)carve((size_t)G * HID * 4);
    float*     cnt    = (float*)carve((size_t)G * 4);
    _Float16*  Wl1h   = (_Float16*)carve((size_t)HID * 64 * 2);
    _Float16*  Wr1h   = (_Float16*)carve((size_t)HID * 64 * 2);
    _Float16*  Wl2h   = (_Float16*)carve((size_t)HID * HID * 2);
    _Float16*  Wr2h   = (_Float16*)carve((size_t)HID * HID * 2);
    _Float16*  Wl3h   = (_Float16*)carve((size_t)HID * HID * 2);
    _Float16*  Wr3h   = (_Float16*)carve((size_t)HID * HID * 2);

    const int TB = 256;
    const int mtiles = Nn / 16;
    const int gemmBlocks = (mtiles + 7) / 8;

    // degree / deg_inv
    hipMemsetAsync(deg, 0, (size_t)Nn * 4, stream);
    k_degree<<<(E + TB - 1) / TB, TB, 0, stream>>>(dst, deg, E);
    k_deginv<<<(Nn + TB - 1) / TB, TB, 0, stream>>>(deg, Nn);

    // weights -> f16 (once; ~160KB total, stays in L2)
    k_cast_f16<<<(HID * 64 + TB - 1) / TB, TB, 0, stream>>>(Wl1, Wl1h, HID * 64);
    k_cast_f16<<<(HID * 64 + TB - 1) / TB, TB, 0, stream>>>(Wr1, Wr1h, HID * 64);
    k_cast_f16<<<(HID * HID + TB - 1) / TB, TB, 0, stream>>>(Wl2, Wl2h, HID * HID);
    k_cast_f16<<<(HID * HID + TB - 1) / TB, TB, 0, stream>>>(Wr2, Wr2h, HID * HID);
    k_cast_f16<<<(HID * HID + TB - 1) / TB, TB, 0, stream>>>(Wl3, Wl3h, HID * HID);
    k_cast_f16<<<(HID * HID + TB - 1) / TB, TB, 0, stream>>>(Wr3, Wr3h, HID * HID);

    // x -> f16 (layer-1 input, ld=64)
    k_cast_f16<<<((Nn * 64) + TB - 1) / TB, TB, 0, stream>>>(x, hA, Nn * 64);

    // layer 1 (K=64): hA -> hB
    hipMemsetAsync(agg, 0, (size_t)Nn * 64 * 4, stream);
    {
        long long tot = (long long)E * 16;
        k_scatter<<<(unsigned)((tot + TB - 1) / TB), TB, 0, stream>>>(hA, src, dst, agg, E, 64, 4);
    }
    k_sage_gemm<64><<<gemmBlocks, TB, 0, stream>>>(hA, agg, deg, Wl1h, bl1, Wr1h, hB, mtiles);

    // layer 2 (K=128): hB -> hA
    hipMemsetAsync(agg, 0, (size_t)Nn * HID * 4, stream);
    {
        long long tot = (long long)E * 32;
        k_scatter<<<(unsigned)((tot + TB - 1) / TB), TB, 0, stream>>>(hB, src, dst, agg, E, 128, 5);
    }
    k_sage_gemm<128><<<gemmBlocks, TB, 0, stream>>>(hB, agg, deg, Wl2h, bl2, Wr2h, hA, mtiles);

    // layer 3 (K=128): hA -> hB
    hipMemsetAsync(agg, 0, (size_t)Nn * HID * 4, stream);
    {
        long long tot = (long long)E * 32;
        k_scatter<<<(unsigned)((tot + TB - 1) / TB), TB, 0, stream>>>(hA, src, dst, agg, E, 128, 5);
    }
    k_sage_gemm<128><<<gemmBlocks, TB, 0, stream>>>(hA, agg, deg, Wl3h, bl3, Wr3h, hB, mtiles);

    // global mean pool + head
    hipMemsetAsync(pooled, 0, (size_t)G * HID * 4, stream);
    hipMemsetAsync(cnt, 0, (size_t)G * 4, stream);
    {
        long long tot = (long long)Nn * 32;
        k_pool<<<(unsigned)((tot + TB - 1) / TB), TB, 0, stream>>>(hB, batch, pooled, Nn);
    }
    k_cnt<<<(Nn + TB - 1) / TB, TB, 0, stream>>>(batch, cnt, Nn);
    k_out<<<(G * TASKS + TB - 1) / TB, TB, 0, stream>>>(pooled, cnt, Wo, bo, out, G);
}
